// MaterialEdgePredictor_50362786513105
// MI455X (gfx1250) — matrix operand order repaired and checked by reading, not compiled, not run on hardware
//
#include <hip/hip_runtime.h>
#include <hip/hip_bf16.h>
#include <stdint.h>

#define H     128
#define NMAT  400000
#define NELEM 118
#define NE    1000000
#define ELAB  400000

typedef __attribute__((ext_vector_type(16))) __bf16 bf16x16;
typedef __attribute__((ext_vector_type(8)))  float  f32x8;

union Frag {
  uint4 q[2];
  unsigned short us[16];
  bf16x16 v;
};

__device__ inline unsigned short f2bf(float x) {
  unsigned u = __float_as_uint(x);
  u += 0x7FFFu + ((u >> 16) & 1u);           // round-to-nearest-even
  return (unsigned short)(u >> 16);
}
__device__ inline float bf2f(unsigned short b) {
  return __uint_as_float(((unsigned)b) << 16);
}
__device__ inline f32x8 zero8() {
  f32x8 z = {0.f,0.f,0.f,0.f,0.f,0.f,0.f,0.f};
  return z;
}
__device__ inline void atom_add_f32(float* p, float v) {
  asm volatile("global_atomic_add_f32 %0, %1, off" :: "v"(p), "v"(v) : "memory");
}
__device__ inline void atom_pk_add_bf16(unsigned* p, unsigned v) {
  asm volatile("global_atomic_pk_add_bf16 %0, %1, off" :: "v"(p), "v"(v) : "memory");
}
// Opaque zero: defeats LICM so LDS loads stay next to their WMMAs (anti-spill).
__device__ inline unsigned opaque_zero() {
  unsigned z = 0;
  asm volatile("" : "+v"(z));
  return z;
}

// ---------------------------------------------------------------- zero fill
__global__ void zero_u32(unsigned* p, long long n) {
  long long stride = (long long)gridDim.x * blockDim.x;
  for (long long i = (long long)blockIdx.x * blockDim.x + threadIdx.x; i < n; i += stride)
    p[i] = 0u;
}

// --------------------------------------------- edge pass: bf16 count matrix
// C[dst, src] += 1.0 (bf16, packed atomic), deg[dst] += 1.0
__global__ void edge_counts(const int* __restrict__ src, const int* __restrict__ dst,
                            unsigned short* Cbf, float* deg) {
  int stride = gridDim.x * blockDim.x;
  for (int e = blockIdx.x * blockDim.x + threadIdx.x; e < NE; e += stride) {
    int s = src[e], d = dst[e];
    size_t elem = ((size_t)d << 7) + (size_t)s;
    unsigned* p = (unsigned*)Cbf + (elem >> 1);
    unsigned val = (s & 1) ? 0x3F800000u : 0x00003F80u;   // bf16 1.0 in hi/lo half
    atom_pk_add_bf16(p, val);
    atom_add_f32(deg + d, 1.0f);
  }
}

// ---------------------- agg_e: gather z_mat rows, reduce into 118 LDS rows
__global__ __launch_bounds__(256) void agg_me_kernel(
    const unsigned short* __restrict__ zmat,
    const int* __restrict__ src, const int* __restrict__ dst,
    float* agg_e, float* cnt_e) {
  __shared__ float acc[NELEM * H];     // 60416 B
  __shared__ float cnt[128];
  const int tid = threadIdx.x;
  for (int i = tid; i < NELEM * H; i += blockDim.x) acc[i] = 0.f;
  if (tid < 128) cnt[tid] = 0.f;
  __syncthreads();
  int lane = tid & 31;
  int gw = (blockIdx.x * blockDim.x + tid) >> 5;
  int nw = (gridDim.x * blockDim.x) >> 5;
  for (int e = gw; e < NE; e += nw) {                 // one edge per wave
    int s = src[e], d = dst[e];
    const unsigned short* p = zmat + ((size_t)s << 7) + lane * 4;
    uint2 q = *(const uint2*)p;                       // 4 bf16 per lane
    float f0 = bf2f((unsigned short)(q.x & 0xFFFF));
    float f1 = bf2f((unsigned short)(q.x >> 16));
    float f2 = bf2f((unsigned short)(q.y & 0xFFFF));
    float f3 = bf2f((unsigned short)(q.y >> 16));
    float* a = acc + d * H + lane * 4;
    __hip_atomic_fetch_add(a + 0, f0, __ATOMIC_RELAXED, __HIP_MEMORY_SCOPE_WORKGROUP);
    __hip_atomic_fetch_add(a + 1, f1, __ATOMIC_RELAXED, __HIP_MEMORY_SCOPE_WORKGROUP);
    __hip_atomic_fetch_add(a + 2, f2, __ATOMIC_RELAXED, __HIP_MEMORY_SCOPE_WORKGROUP);
    __hip_atomic_fetch_add(a + 3, f3, __ATOMIC_RELAXED, __HIP_MEMORY_SCOPE_WORKGROUP);
    if (lane == 0)
      __hip_atomic_fetch_add(cnt + d, 1.0f, __ATOMIC_RELAXED, __HIP_MEMORY_SCOPE_WORKGROUP);
  }
  __syncthreads();
  for (int i = tid; i < NELEM * H; i += blockDim.x) atom_add_f32(agg_e + i, acc[i]);
  if (tid < NELEM) atom_add_f32(cnt_e + tid, cnt[tid]);
}

// ----------------------------------------- element-side chain (tiny, 1 block)
__global__ __launch_bounds__(256) void elem_chain(
    const float* __restrict__ x_elem, const float* __restrict__ Wp, const float* __restrict__ bp,
    const float* __restrict__ agg_e, const float* __restrict__ cnt_e,
    const float* __restrict__ Wl, const float* __restrict__ Wr, const float* __restrict__ bme,
    const float* __restrict__ Wf, const float* __restrict__ bfe,
    const float* __restrict__ Wo, const float* __restrict__ bo,
    float* zelem_pad, unsigned short* zelem4) {
  __shared__ float z[NELEM * H];
  __shared__ float t[NELEM * H];
  const int tid = threadIdx.x, BS = blockDim.x;
  for (int idx = tid; idx < NELEM * H; idx += BS) {   // z_elem = x_elem@Wp + bp
    int i = idx >> 7, j = idx & 127;
    float s = bp[j];
    for (int k = 0; k < H; k++) s += x_elem[i * H + k] * Wp[k * H + j];
    z[idx] = s;
  }
  __syncthreads();
  for (int idx = tid; idx < H * H; idx += BS)         // padded f32 copy for WMMA B
    zelem_pad[idx] = (idx < NELEM * H) ? z[idx] : 0.f;
  for (int idx = tid; idx < NELEM * H; idx += BS) {   // z2 = relu(mean@Wl + z@Wr + b)
    int i = idx >> 7, j = idx & 127;
    float inv = 1.f / fmaxf(cnt_e[i], 1.f);
    float s = bme[j];
    for (int k = 0; k < H; k++)
      s += agg_e[i * H + k] * inv * Wl[k * H + j] + z[i * H + k] * Wr[k * H + j];
    t[idx] = fmaxf(s, 0.f);
  }
  __syncthreads();
  for (int idx = tid; idx < NELEM * H; idx += BS) {   // z3 = relu(z2@Wf + bf)
    int i = idx >> 7, j = idx & 127;
    float s = bfe[j];
    for (int k = 0; k < H; k++) s += t[i * H + k] * Wf[k * H + j];
    z[idx] = fmaxf(s, 0.f);
  }
  __syncthreads();
  for (int idx = tid; idx < H * H; idx += BS) {       // z4 = z3@Wo + bo -> bf16
    if (idx < NELEM * H) {
      int i = idx >> 7, j = idx & 127;
      float s = bo[j];
      for (int k = 0; k < H; k++) s += z[i * H + k] * Wo[k * H + j];
      zelem4[idx] = f2bf(s);
    } else zelem4[idx] = 0;
  }
}

// ---------------------------------------------------------- main WMMA GEMM
// out[M,128](bf16) = epilogue( A1@B1 (+ A2@B2) ); A row-major [M,K], B f32 [K,128]
// EPI: 0 = +bias, 1 = +bias+relu, 2 = *1/max(deg,1)
template <int K, bool AF32, bool DUAL, int EPI>
__global__ __launch_bounds__(256) void gemm128(
    const void* __restrict__ A1_, const float* __restrict__ B1f,
    const void* __restrict__ A2_, const float* __restrict__ B2f,
    const float* __restrict__ bias, const float* __restrict__ deg,
    unsigned short* __restrict__ out, int M) {
  constexpr int KT = K / 32;
  __shared__ unsigned short Bsw[(DUAL ? 2 : 1) * KT * 8 * 32 * 16];
  for (int idx = threadIdx.x; idx < K * 128; idx += blockDim.x) {  // swizzle B -> LDS bf16
    int k = idx >> 7, n = idx & 127;
    int kt = k >> 5, kr = k & 31, nt = n >> 4;
    int ln = ((kr >> 3) & 1) * 16 + (n & 15);
    int slot = (kr & 7) + ((kr >> 4) << 3);
    int pos = (((kt * 8 + nt) * 32 + ln) << 4) + slot;
    Bsw[pos] = f2bf(B1f[idx]);
    if (DUAL) Bsw[KT * 8 * 32 * 16 + pos] = f2bf(B2f[idx]);
  }
  __syncthreads();
  int lane = threadIdx.x & 31, wave = threadIdx.x >> 5;
  int row0 = blockIdx.x * 128 + wave * 16;
  int r = lane & 15, half = lane >> 4;
  f32x8 acc[8];
#pragma unroll
  for (int nt = 0; nt < 8; nt++) acc[nt] = zero8();
  const int NPASS = DUAL ? 2 : 1;
#pragma unroll
  for (int pass = 0; pass < NPASS; pass++) {
    const void* Ap = pass ? A2_ : A1_;
    const unsigned short* BL = Bsw + pass * KT * 8 * 32 * 16;
#pragma unroll
    for (int kt = 0; kt < KT; kt++) {
      Frag a;
      if (AF32) {
        const float* ap = (const float*)Ap + (size_t)(row0 + r) * K + kt * 32 + half * 8;
#pragma unroll
        for (int i = 0; i < 8; i++) { a.us[i] = f2bf(ap[i]); a.us[8 + i] = f2bf(ap[16 + i]); }
      } else {
        const unsigned short* ap =
            (const unsigned short*)Ap + (size_t)(row0 + r) * K + kt * 32 + half * 8;
        a.q[0] = *(const uint4*)ap;
        a.q[1] = *(const uint4*)(ap + 16);
      }
#pragma unroll
      for (int nt = 0; nt < 8; nt++) {
        Frag b;
        const uint4* bp = (const uint4*)(BL + (((kt * 8 + nt) * 32 + lane) << 4));
        b.q[0] = bp[0]; b.q[1] = bp[1];
        acc[nt] = __builtin_amdgcn_wmma_f32_16x16x32_bf16(
            false, a.v, false, b.v, (short)0, acc[nt], false, false);
      }
    }
  }
#pragma unroll
  for (int nt = 0; nt < 8; nt++) {
    int col = nt * 16 + r;
#pragma unroll
    for (int i = 0; i < 8; i++) {
      int row = row0 + i + half * 8;
      float v = acc[nt][i];
      if constexpr (EPI == 0 || EPI == 1) v += bias[col];
      if constexpr (EPI == 1) v = fmaxf(v, 0.f);
      if constexpr (EPI == 2) v *= 1.f / fmaxf(deg[row], 1.f);
      out[(size_t)row * 128 + col] = f2bf(v);
    }
  }
}

// ------------------------------------------------------------ edge decoder
__global__ __launch_bounds__(256) void decoder_kernel(
    const unsigned short* __restrict__ zmat4, const unsigned short* __restrict__ zelem4,
    const int* __restrict__ lab_src, const int* __restrict__ lab_dst,
    const float* __restrict__ W1, const float* __restrict__ b1,
    const float* __restrict__ W2, const float* __restrict__ b2,
    float* __restrict__ out) {
  __shared__ unsigned short Bsw[8 * 16 * 32 * 16];   // 128 KB: W1 swizzled bf16
  __shared__ float w2s[256];
  __shared__ float b1s[256];
  for (int idx = threadIdx.x; idx < 256 * 256; idx += blockDim.x) {
    int k = idx >> 8, n = idx & 255;
    int kt = k >> 5, kr = k & 31, nt = n >> 4;
    int ln = ((kr >> 3) & 1) * 16 + (n & 15);
    int slot = (kr & 7) + ((kr >> 4) << 3);
    Bsw[(((kt * 16 + nt) * 32 + ln) << 4) + slot] = f2bf(W1[idx]);
  }
  if (threadIdx.x < 256) { w2s[threadIdx.x] = W2[threadIdx.x]; b1s[threadIdx.x] = b1[threadIdx.x]; }
  __syncthreads();
  float b2v = b2[0];
  int lane = threadIdx.x & 31, wave = threadIdx.x >> 5;
  int r = lane & 15, half = lane >> 4;
  for (int tile = blockIdx.x * 8 + wave; tile < ELAB / 16; tile += gridDim.x * 8) {
    int e0 = tile * 16;
    int src = lab_src[e0 + r];
    int dst = lab_dst[e0 + r];
    const unsigned short* mrow = zmat4 + (size_t)src * 128;
    const unsigned short* erow = zelem4 + (size_t)dst * 128;
    Frag a[8];
#pragma unroll
    for (int kt = 0; kt < 8; kt++) {                  // K=256: mat cols then elem cols
      const unsigned short* base = (kt < 4 ? mrow + kt * 32 : erow + (kt - 4) * 32) + half * 8;
      a[kt].q[0] = *(const uint4*)base;
      a[kt].q[1] = *(const uint4*)(base + 16);
    }
    // Laundered zero offset: makes LDS addresses loop-variant so the B-tile
    // ds_loads are NOT hoisted out of the tile loop (which caused 1024-dword
    // scratch spills/reloads in front of every WMMA).
    unsigned bz = opaque_zero();
    const unsigned short* Bdyn = Bsw + bz;
    const float* w2d = w2s + bz;
    const float* b1d = b1s + bz;
    float plog[8];
#pragma unroll
    for (int i = 0; i < 8; i++) plog[i] = 0.f;
    for (int nt = 0; nt < 16; nt++) {
      f32x8 acc = zero8();
#pragma unroll
      for (int kt = 0; kt < 8; kt++) {
        Frag b;
        const uint4* bp = (const uint4*)(Bdyn + (((kt * 16 + nt) * 32 + lane) << 4));
        b.q[0] = bp[0]; b.q[1] = bp[1];
        acc = __builtin_amdgcn_wmma_f32_16x16x32_bf16(
            false, a[kt].v, false, b.v, (short)0, acc, false, false);
      }
      int col = nt * 16 + r;
      float wv = w2d[col], bb = b1d[col];
#pragma unroll
      for (int i = 0; i < 8; i++) plog[i] += fmaxf(acc[i] + bb, 0.f) * wv;  // fold W2 head
    }
#pragma unroll
    for (int i = 0; i < 8; i++) {                     // reduce over 16 N-lanes (wave32)
      float v = plog[i];
      v += __shfl_xor(v, 1, 32);
      v += __shfl_xor(v, 2, 32);
      v += __shfl_xor(v, 4, 32);
      v += __shfl_xor(v, 8, 32);
      plog[i] = v;
    }
    if (r == 0) {
#pragma unroll
      for (int i = 0; i < 8; i++) {
        int row = i + half * 8;
        float logit = plog[i] + b2v;
        out[e0 + row] = 1.f / (1.f + __expf(-logit));
      }
    }
  }
}

// ------------------------------------------------------------------- launch
extern "C" void kernel_launch(void* const* d_in, const int* in_sizes, int n_in,
                              void* d_out, int out_size, void* d_ws, size_t ws_size,
                              hipStream_t stream) {
  (void)in_sizes; (void)n_in; (void)out_size; (void)ws_size;
  const float* x_mat  = (const float*)d_in[0];
  const float* x_elem = (const float*)d_in[1];
  const int* src_me = (const int*)d_in[2];
  const int* dst_me = (const int*)d_in[3];
  const int* src_em = (const int*)d_in[4];
  const int* dst_em = (const int*)d_in[5];
  const int* lab_src = (const int*)d_in[6];
  const int* lab_dst = (const int*)d_in[7];
  const float* Wp_mat = (const float*)d_in[8];  const float* bp_mat = (const float*)d_in[9];
  const float* Wp_elem = (const float*)d_in[10]; const float* bp_elem = (const float*)d_in[11];
  const float* Wl_me = (const float*)d_in[12]; const float* Wr_me = (const float*)d_in[13];
  const float* b_me  = (const float*)d_in[14];
  const float* Wl_em = (const float*)d_in[15]; const float* Wr_em = (const float*)d_in[16];
  const float* b_em  = (const float*)d_in[17];
  const float* Wf_mat = (const float*)d_in[18]; const float* bf_mat = (const float*)d_in[19];
  const float* Wf_elem = (const float*)d_in[20]; const float* bf_elem = (const float*)d_in[21];
  const float* Wo_mat = (const float*)d_in[22]; const float* bo_mat = (const float*)d_in[23];
  const float* Wo_elem = (const float*)d_in[24]; const float* bo_elem = (const float*)d_in[25];
  const float* W1 = (const float*)d_in[26]; const float* b1 = (const float*)d_in[27];
  const float* W2 = (const float*)d_in[28]; const float* b2 = (const float*)d_in[29];
  float* out = (float*)d_out;

  const size_t SZ = (size_t)NMAT * H * 2;             // one bf16 [400000,128] buffer
  char* w = (char*)d_ws;
  unsigned short* B0 = (unsigned short*)w;            // z_mat -> z_mat3
  unsigned short* B1 = (unsigned short*)(w + SZ);     // counts C -> z_mat2
  unsigned short* B2 = (unsigned short*)(w + 2 * SZ); // agg_m -> z_mat4
  float* deg   = (float*)(w + 3 * SZ);                // [NMAT]
  float* agg_e = deg + NMAT;                          // [118*128]
  float* cnt_e = agg_e + NELEM * H;                   // [128]
  float* zelem_pad = cnt_e + 128;                     // [128*128] f32
  unsigned short* zelem4 = (unsigned short*)(zelem_pad + H * H); // [128*128] bf16

  // zero: count matrix, deg + agg_e + cnt_e (contiguous)
  zero_u32<<<2048, 256, 0, stream>>>((unsigned*)B1, (long long)NMAT * H / 2);
  zero_u32<<<256, 256, 0, stream>>>((unsigned*)deg, (long long)(NMAT + NELEM * H + 128));
  // z_mat = x_mat @ Wp_mat + bp_mat  (f32 A, K=256) -> B0 bf16
  gemm128<256, true, false, 0><<<NMAT / 128, 256, 0, stream>>>(
      x_mat, Wp_mat, nullptr, nullptr, bp_mat, nullptr, B0, NMAT);
  // C[dst,src] += 1 (pk bf16 atomic), deg[dst] += 1
  edge_counts<<<2048, 256, 0, stream>>>(src_em, dst_em, B1, deg);
  // agg_e: LDS-accumulated gather-reduce of z_mat over me edges
  agg_me_kernel<<<120, 256, 0, stream>>>(B0, src_me, dst_me, agg_e, cnt_e);
  // element-side chain (tiny), emits zelem_pad f32 and zelem4 bf16
  elem_chain<<<1, 256, 0, stream>>>(x_elem, Wp_elem, bp_elem, agg_e, cnt_e,
                                    Wl_me, Wr_me, b_me, Wf_elem, bf_elem,
                                    Wo_elem, bo_elem, zelem_pad, zelem4);
  // agg_m = (C @ z_elem_pad) / max(deg,1) -> B2 bf16
  gemm128<128, false, false, 2><<<NMAT / 128, 256, 0, stream>>>(
      B1, zelem_pad, nullptr, nullptr, nullptr, deg, B2, NMAT);
  // z_mat2 = relu(agg_m@Wl_em + z_mat@Wr_em + b_em) -> B1 (dual GEMM)
  gemm128<128, false, true, 1><<<NMAT / 128, 256, 0, stream>>>(
      B2, Wl_em, B0, Wr_em, b_em, nullptr, B1, NMAT);
  // z_mat3 = relu(z_mat2@Wf_mat + bf_mat) -> B0
  gemm128<128, false, false, 1><<<NMAT / 128, 256, 0, stream>>>(
      B1, Wf_mat, nullptr, nullptr, bf_mat, nullptr, B0, NMAT);
  // z_mat4 = z_mat3@Wo_mat + bo_mat -> B2
  gemm128<128, false, false, 0><<<NMAT / 128, 256, 0, stream>>>(
      B0, Wo_mat, nullptr, nullptr, bo_mat, nullptr, B2, NMAT);
  // edge decoder: gather + [zcat@W1+b1]+relu, fold W2 head, sigmoid
  decoder_kernel<<<256, 256, 0, stream>>>(B2, zelem4, lab_src, lab_dst,
                                          W1, b1, W2, b2, out);
}
